// LearnedDoubleCrossBabyWithLearnedSinEmbedding_73486890435290
// MI455X (gfx1250) — compile-verified, matrix operand order checked by reading, not run on hardware
//
#include <hip/hip_runtime.h>
#include <hip/hip_bf16.h>
#include <math.h>

// ---------------------------------------------------------------------------
// Problem constants: B=8, L=128, V=32000, D=64, R1=128, R2=320, UP=640
// ---------------------------------------------------------------------------
constexpr int kB  = 8;
constexpr int kL  = 128;
constexpr int kV  = 32000;
constexpr int kD  = 64;
constexpr int kR1 = 128;
constexpr int kR2 = 320;
constexpr int kUP = 640;

typedef __attribute__((ext_vector_type(16))) _Float16 v16h;
typedef __attribute__((ext_vector_type(8)))  _Float16 v8h;
typedef __attribute__((ext_vector_type(8)))  float    v8f;

// --- fragment loaders for v_wmma_f32_16x16x32_f16 --------------------------
// p points at (row base + k0 + (lane>>4)*8).
// element e<8 -> K = k0 + half*8 + e ; e>=8 -> K = k0 + 16 + half*8 + (e-8)
__device__ __forceinline__ v16h frag_ld_f32(const float* __restrict__ p) {
  v16h f;
#pragma unroll
  for (int e = 0; e < 8; ++e) f[e] = (_Float16)p[e];
#pragma unroll
  for (int e = 0; e < 8; ++e) f[e + 8] = (_Float16)p[e + 16];
  return f;
}

__device__ __forceinline__ v16h frag_ld_f16(const _Float16* p) {
  v8h lo = *(const v8h*)p;          // one b128 load
  v8h hi = *(const v8h*)(p + 16);   // one b128 load
  return __builtin_shufflevector(lo, hi, 0, 1, 2, 3, 4, 5, 6, 7,
                                         8, 9, 10, 11, 12, 13, 14, 15);
}

// ---------------------------------------------------------------------------
// Generic C = relu(A @ W^T + bias). A:[M,K], W:[N,K] row-major, f32.
// One wave -> 16 x (16*NT) strip (A fragment reused NT times per k-step).
// Requires N % (16*NT) == 0, K % 32 == 0, M % 16 == 0.
// ---------------------------------------------------------------------------
template <int NT>
__global__ void relu_gemm_nt_wmma(const float* __restrict__ A,
                                  const float* __restrict__ W,
                                  const float* __restrict__ bias,
                                  float* __restrict__ C,
                                  int M, int N, int K,
                                  long sA, long sW, long sC) {
  const int b = blockIdx.z;
  A += (long)b * sA;  W += (long)b * sW;  C += (long)b * sC;

  const int wave = (int)((blockIdx.x * blockDim.x + threadIdx.x) >> 5);
  const int lane = (int)(threadIdx.x & 31);
  const int strips = N / (16 * NT);
  const int total  = (M >> 4) * strips;
  if (wave >= total) return;                 // wave-uniform

  const int mt = wave / strips;
  const int st = wave % strips;
  const int l16  = lane & 15;
  const int half = lane >> 4;
  const int m = (mt << 4) + l16;

  const float* ap = A + (long)m * K + (half << 3);
  const float* wp[NT];
  float bn[NT];
#pragma unroll
  for (int t = 0; t < NT; ++t) {
    const int n = st * (16 * NT) + t * 16 + l16;
    wp[t] = W + (long)n * K + (half << 3);
    bn[t] = bias[n];
  }

  v8f acc[NT] = {};
  for (int k0 = 0; k0 < K; k0 += 32) {
    v16h af = frag_ld_f32(ap + k0);
#pragma unroll
    for (int t = 0; t < NT; ++t) {
      v16h wf = frag_ld_f32(wp[t] + k0);
      acc[t] = __builtin_amdgcn_wmma_f32_16x16x32_f16(
          false, af, false, wf, (short)0, acc[t], false, false);
    }
  }

#pragma unroll
  for (int t = 0; t < NT; ++t) {
    const int n = st * (16 * NT) + t * 16 + l16;
#pragma unroll
    for (int r = 0; r < 8; ++r) {
      const int row = (mt << 4) + r + (half << 3);
      float v = acc[t][r] + bn[t];
      v = v > 0.0f ? v : 0.0f;
      C[(long)row * N + n] = v;
    }
  }
}

// ---------------------------------------------------------------------------
// Split-K GEMM with atomic accumulation (memory-bound embedding GEMM:
// M=1024, N=64, K=32000 -> grid.y slices K so >1000 waves stream x).
// Cacc must be pre-zeroed; bias+relu applied by a later pass.
// ---------------------------------------------------------------------------
template <int NT>
__global__ void gemm_nt_splitk_wmma(const float* __restrict__ A,
                                    const float* __restrict__ W,
                                    float* __restrict__ Cacc,
                                    int M, int N, int K, int kChunk) {
  const int ks = blockIdx.y;
  const int kbeg = ks * kChunk;
  const int kend = (kbeg + kChunk < K) ? (kbeg + kChunk) : K;

  const int wave = (int)((blockIdx.x * blockDim.x + threadIdx.x) >> 5);
  const int lane = (int)(threadIdx.x & 31);
  const int strips = N / (16 * NT);
  const int total  = (M >> 4) * strips;
  if (wave >= total) return;

  const int mt = wave / strips;
  const int st = wave % strips;
  const int l16  = lane & 15;
  const int half = lane >> 4;
  const int m = (mt << 4) + l16;

  const float* ap = A + (long)m * K + (half << 3);
  const float* wp[NT];
#pragma unroll
  for (int t = 0; t < NT; ++t) {
    const int n = st * (16 * NT) + t * 16 + l16;
    wp[t] = W + (long)n * K + (half << 3);
  }

  v8f acc[NT] = {};
  for (int k0 = kbeg; k0 < kend; k0 += 32) {
    v16h af = frag_ld_f32(ap + k0);
#pragma unroll
    for (int t = 0; t < NT; ++t) {
      v16h wf = frag_ld_f32(wp[t] + k0);
      acc[t] = __builtin_amdgcn_wmma_f32_16x16x32_f16(
          false, af, false, wf, (short)0, acc[t], false, false);
    }
  }

#pragma unroll
  for (int t = 0; t < NT; ++t) {
    const int n = st * (16 * NT) + t * 16 + l16;
#pragma unroll
    for (int r = 0; r < 8; ++r) {
      const int row = (mt << 4) + r + (half << 3);
      atomicAdd(&Cacc[(long)row * N + n], acc[t][r]);
    }
  }
}

// ---------------------------------------------------------------------------
// Cross-term factorization: Mout[b, r*DW + j] = sum_k spad[b,k] * Wr[r, k*DW+j]
// spad is a ZERO-PADDED [16, K] copy of the row-sums (rows 8..15 are zero):
// unconditional lane loads -> no EXEC divergence; DW templated so the
// k-strided B-column addresses are immediate offsets.
// ---------------------------------------------------------------------------
template <int DW>
__global__ void cross_reduce_wmma(const float* __restrict__ spad, // [16, K]
                                  const float* __restrict__ Wr,   // [R, K*DW]
                                  float* __restrict__ Mout,       // [kB, Ncols]
                                  int K, int Ncols) {
  const int wave = (int)((blockIdx.x * blockDim.x + threadIdx.x) >> 5);
  const int lane = (int)(threadIdx.x & 31);
  const int ntiles = Ncols >> 4;
  if (wave >= ntiles) return;

  const int l16  = lane & 15;
  const int half = lane >> 4;
  const int n = (wave << 4) + l16;
  const int r = n / DW;
  const int j = n % DW;

  const float* wcol = Wr + (long)r * K * DW + j;            // k stride = DW
  const float* srow = spad + (long)l16 * K + (half << 3);   // padded: no guard

  v8f acc = {};
  for (int k0 = 0; k0 < K; k0 += 32) {
    v16h af = frag_ld_f32(srow + k0);
    const float* wk = wcol + (long)(k0 + (half << 3)) * DW;
    v16h wf;
#pragma unroll
    for (int e = 0; e < 16; ++e) {
      const int kk = (e & 7) + ((e >> 3) << 4);             // compile-time
      wf[e] = (_Float16)wk[kk * DW];                        // imm offsets
    }
    acc = __builtin_amdgcn_wmma_f32_16x16x32_f16(
        false, af, false, wf, (short)0, acc, false, false);
  }

#pragma unroll
  for (int r8 = 0; r8 < 8; ++r8) {
    const int row = r8 + (half << 3);
    if (row < kB) Mout[(long)row * Ncols + n] = acc[r8];
  }
}

// ---------------------------------------------------------------------------
// Fused final stage with LDS staging, all-f16 operands:
//   out[b,v] = relu( sum_l relu(h4[b,l,:]@W_up2[v,:] + b_up2[v]) * W_fin[l]
//                    + b_fin )
// One workgroup (8 waves) = one b + 8 consecutive v-tiles; h4[b] staged in LDS.
// k0 outer / lt inner with 8 accumulators; A and W fragments are explicitly
// double-buffered (a_cur/a_nxt, wf/wf_nxt) so no register range is both the
// operand of an in-flight WMMA and the target of the next load -> no WAR
// hazard NOPs, ds/global latency overlaps the WMMA issue stream.
// ---------------------------------------------------------------------------
constexpr int kLdsRow = kUP + 8;                  // pad 8 halves -> bank spread

__global__ void up2_fin_lds_wmma(const _Float16* __restrict__ h4f, // [B,L,UP]
                                 const _Float16* __restrict__ Wuf, // [V,UP]
                                 const float* __restrict__ bu,     // [V]
                                 const float* __restrict__ wfin,   // [L]
                                 const float* __restrict__ bfin,   // [1]
                                 float* __restrict__ out) {        // [B,V]
  extern __shared__ _Float16 lds[];               // kL * kLdsRow halves
  const int b   = blockIdx.y;
  const int tid = (int)threadIdx.x;

  const _Float16* src = h4f + (long)b * kL * kUP;
  for (int idx = tid; idx < kL * (kUP / 8); idx += 256) {
    const int row = idx / (kUP / 8);
    const int c8  = idx % (kUP / 8);
    *(v8h*)(lds + (long)row * kLdsRow + c8 * 8) =
        *(const v8h*)(src + (long)row * kUP + c8 * 8);
  }
  __syncthreads();

  const int wave = tid >> 5;
  const int lane = tid & 31;
  const int vt = blockIdx.x * 8 + wave;           // exact cover: 250*8 = 2000
  const int l16  = lane & 15;
  const int half = lane >> 4;
  const int n = (vt << 4) + l16;

  const _Float16* wp  = Wuf + (long)n * kUP + (half << 3);
  const _Float16* ab  = lds + (long)l16 * kLdsRow + (half << 3);
  const float bn = bu[n];

  constexpr int LT = kL / 16;                     // 8 row tiles
  v8f c[LT] = {};                                 // 64 VGPRs of accumulators

  v16h wf = frag_ld_f16(wp);                      // prefetch W for k0 = 0
  for (int k0 = 0; k0 < kUP; k0 += 32) {
    v16h wf_nxt = wf;
    if (k0 + 32 < kUP) wf_nxt = frag_ld_f16(wp + k0 + 32);  // hide global lat.

    v16h a_cur = frag_ld_f16(ab + k0);            // lt = 0
#pragma unroll
    for (int lt = 0; lt < LT; ++lt) {
      v16h a_nxt = a_cur;
      if (lt + 1 < LT)                            // load ahead of the wmma
        a_nxt = frag_ld_f16(ab + (long)(lt + 1) * 16 * kLdsRow + k0);
      c[lt] = __builtin_amdgcn_wmma_f32_16x16x32_f16(
          false, a_cur, false, wf, (short)0, c[lt], false, false);
      a_cur = a_nxt;
    }
    wf = wf_nxt;
  }

  float red = 0.0f;
#pragma unroll
  for (int lt = 0; lt < LT; ++lt) {
#pragma unroll
    for (int r = 0; r < 8; ++r) {
      float v = c[lt][r] + bn;
      v = v > 0.0f ? v : 0.0f;
      red += v * wfin[(lt << 4) + r + (half << 3)];
    }
  }

  red += __shfl_xor(red, 16, 32);                 // combine row halves
  if (half == 0) {
    float v = red + bfin[0];
    out[(long)b * kV + n] = v > 0.0f ? v : 0.0f;
  }
}

// ---------------------------------------------------------------------------
// Small helpers
// ---------------------------------------------------------------------------
__global__ void pos_enc_kernel(float* __restrict__ P, int seq, int d) {
  const int idx = (int)(blockIdx.x * blockDim.x + threadIdx.x);
  const int hd = d >> 1;
  if (idx >= seq * hd) return;
  const int k = idx / hd;
  const int i = idx % hd;
  const float ang = (float)k * __expf(-(2.0f * (float)i / (float)d) * __logf(10000.0f));
  P[(long)k * d + 2 * i]     = __sinf(ang);
  P[(long)k * d + 2 * i + 1] = __cosf(ang);
}

__global__ void rowsum_kernel(const float* __restrict__ h, float* __restrict__ s,
                              int rows, int ncols) {
  const int r = (int)(blockIdx.x * blockDim.x + threadIdx.x);
  if (r >= rows) return;
  const float* p = h + (long)r * ncols;
  float acc = 0.0f;
  for (int i = 0; i < ncols; ++i) acc += p[i];
  s[r] = acc;
}

__global__ void add_pos_kernel(float* __restrict__ h, const float* __restrict__ pos,
                               int total, int LN) {
  const int i = (int)(blockIdx.x * blockDim.x + threadIdx.x);
  if (i < total) h[i] += pos[i % LN];
}

__global__ void zero_kernel(float* __restrict__ p, int n) {
  const int i = (int)(blockIdx.x * blockDim.x + threadIdx.x);
  if (i < n) p[i] = 0.0f;
}

__global__ void bias_relu_kernel(float* __restrict__ h, const float* __restrict__ bias,
                                 int total, int N) {
  const int i = (int)(blockIdx.x * blockDim.x + threadIdx.x);
  if (i >= total) return;
  float v = h[i] + bias[i % N];
  h[i] = v > 0.0f ? v : 0.0f;
}

__global__ void cvt_f32_f16_kernel(const float* __restrict__ in,
                                   _Float16* __restrict__ out, long n) {
  const long i = (long)blockIdx.x * blockDim.x + threadIdx.x;
  if (i < n) out[i] = (_Float16)in[i];
}

// ---------------------------------------------------------------------------
// Orchestration
// ---------------------------------------------------------------------------
extern "C" void kernel_launch(void* const* d_in, const int* in_sizes, int n_in,
                              void* d_out, int out_size, void* d_ws, size_t ws_size,
                              hipStream_t stream) {
  (void)in_sizes; (void)n_in; (void)out_size; (void)ws_size;

  const float* x      = (const float*)d_in[0];
  const float* W_emb  = (const float*)d_in[1];
  const float* b_emb  = (const float*)d_in[2];
  const float* W_pos1 = (const float*)d_in[3];
  const float* b_pos1 = (const float*)d_in[4];
  const float* W_red  = (const float*)d_in[5];
  const float* b_red  = (const float*)d_in[6];
  const float* W_pos2 = (const float*)d_in[7];
  const float* b_pos2 = (const float*)d_in[8];
  const float* W_red2 = (const float*)d_in[9];
  const float* b_red2 = (const float*)d_in[10];
  const float* W_pos3 = (const float*)d_in[11];
  const float* b_pos3 = (const float*)d_in[12];
  const float* W_up1  = (const float*)d_in[13];
  const float* b_up1  = (const float*)d_in[14];
  const float* W_up2  = (const float*)d_in[15];
  const float* b_up2  = (const float*)d_in[16];
  const float* W_fin  = (const float*)d_in[17];
  const float* b_fin  = (const float*)d_in[18];
  float* out = (float*)d_out;

  // Workspace carve-up (float units)
  float* ws = (float*)d_ws;
  size_t off = 0;
  float* P1    = ws + off; off += (size_t)kL * (2 * kD);
  float* P2    = ws + off; off += (size_t)kL * (2 * kR1);
  float* P3    = ws + off; off += (size_t)kL * (2 * kR2);
  float* h1    = ws + off; off += (size_t)kB * kL * kD;     // emb acc + result
  float* pos1o = ws + off; off += (size_t)kL * kD;
  float* spad1 = ws + off; off += (size_t)16 * kL;          // zero-padded sums
  float* M1    = ws + off; off += (size_t)kB * kR1 * kD;
  float* h2    = ws + off; off += (size_t)kB * kL * kR1;
  float* pos2o = ws + off; off += (size_t)kL * kR1;
  float* spad2 = ws + off; off += (size_t)16 * kL;
  float* M2    = ws + off; off += (size_t)kB * kR2 * kR1;
  float* h3    = ws + off; off += (size_t)kB * kL * kR2;
  float* pos3o = ws + off; off += (size_t)kL * kR2;
  float* h4    = ws + off; off += (size_t)kB * kL * kUP;
  _Float16* Wuf = (_Float16*)(ws + off); off += ((size_t)kV * kUP + 1) / 2;
  _Float16* h4f = (_Float16*)(ws + off); off += ((size_t)kB * kL * kUP + 1) / 2;

  const int TB = 256;                               // 8 waves / block
  auto wave_blocks = [](int waves) { return (waves + 7) / 8; };

  // f16 copy of W_up2 (used by the final fused GEMM)
  {
    const long n = (long)kV * kUP;
    cvt_f32_f16_kernel<<<(unsigned)((n + TB - 1) / TB), TB, 0, stream>>>(W_up2, Wuf, n);
  }

  // Positional encodings
  pos_enc_kernel<<<(kL * kD  + TB - 1) / TB, TB, 0, stream>>>(P1, kL, 2 * kD);
  pos_enc_kernel<<<(kL * kR1 + TB - 1) / TB, TB, 0, stream>>>(P2, kL, 2 * kR1);
  pos_enc_kernel<<<(kL * kR2 + TB - 1) / TB, TB, 0, stream>>>(P3, kL, 2 * kR2);

  // zero: emb accumulator + the padded rows of both spads
  zero_kernel<<<(kB * kL * kD + TB - 1) / TB, TB, 0, stream>>>(h1, kB * kL * kD);
  zero_kernel<<<(16 * kL + TB - 1) / TB, TB, 0, stream>>>(spad1, 16 * kL);
  zero_kernel<<<(16 * kL + TB - 1) / TB, TB, 0, stream>>>(spad2, 16 * kL);

  // h1 = relu(x @ W_emb^T + b_emb): split-K (16 slices) + atomic accumulate
  gemm_nt_splitk_wmma<4><<<dim3(wave_blocks((kB * kL / 16) * (kD / 64)), 16, 1),
                           TB, 0, stream>>>(x, W_emb, h1, kB * kL, kD, kV, 2048);
  bias_relu_kernel<<<(kB * kL * kD + TB - 1) / TB, TB, 0, stream>>>(
      h1, b_emb, kB * kL * kD, kD);

  // pos1o = relu(P1 @ W_pos1^T + b_pos1); h1 += pos1o
  relu_gemm_nt_wmma<4><<<dim3(wave_blocks((kL / 16) * (kD / 64)), 1, 1), TB, 0, stream>>>(
      P1, W_pos1, b_pos1, pos1o, kL, kD, 2 * kD, 0, 0, 0);
  add_pos_kernel<<<(kB * kL * kD + TB - 1) / TB, TB, 0, stream>>>(
      h1, pos1o, kB * kL * kD, kL * kD);

  // s1 (rows 0..7 of spad1), M1, h2
  rowsum_kernel<<<(kB * kL + TB - 1) / TB, TB, 0, stream>>>(h1, spad1, kB * kL, kD);
  cross_reduce_wmma<kD><<<dim3(wave_blocks((kR1 * kD) / 16)), TB, 0, stream>>>(
      spad1, W_red, M1, kL, kR1 * kD);
  relu_gemm_nt_wmma<4><<<dim3(wave_blocks((kL / 16) * (kR1 / 64)), 1, kB), TB, 0, stream>>>(
      h1, M1, b_red, h2, kL, kR1, kD,
      (long)kL * kD, (long)kR1 * kD, (long)kL * kR1);

  // pos2o; h2 += pos2o
  relu_gemm_nt_wmma<4><<<dim3(wave_blocks((kL / 16) * (kR1 / 64)), 1, 1), TB, 0, stream>>>(
      P2, W_pos2, b_pos2, pos2o, kL, kR1, 2 * kR1, 0, 0, 0);
  add_pos_kernel<<<(kB * kL * kR1 + TB - 1) / TB, TB, 0, stream>>>(
      h2, pos2o, kB * kL * kR1, kL * kR1);

  // s2, M2, h3
  rowsum_kernel<<<(kB * kL + TB - 1) / TB, TB, 0, stream>>>(h2, spad2, kB * kL, kR1);
  cross_reduce_wmma<kR1><<<dim3(wave_blocks((kR2 * kR1) / 16)), TB, 0, stream>>>(
      spad2, W_red2, M2, kL, kR2 * kR1);
  relu_gemm_nt_wmma<4><<<dim3(wave_blocks((kL / 16) * (kR2 / 64)), 1, kB), TB, 0, stream>>>(
      h2, M2, b_red2, h3, kL, kR2, kR1,
      (long)kL * kR1, (long)kR2 * kR1, (long)kL * kR2);

  // pos3o; h3 += pos3o
  relu_gemm_nt_wmma<4><<<dim3(wave_blocks((kL / 16) * (kR2 / 64)), 1, 1), TB, 0, stream>>>(
      P3, W_pos3, b_pos3, pos3o, kL, kR2, 2 * kR2, 0, 0, 0);
  add_pos_kernel<<<(kB * kL * kR2 + TB - 1) / TB, TB, 0, stream>>>(
      h3, pos3o, kB * kL * kR2, kL * kR2);

  // h4 = relu(h3 @ W_up1^T + b_up1); then f16 copy
  relu_gemm_nt_wmma<4><<<dim3(wave_blocks((kB * kL / 16) * (kUP / 64)), 1, 1), TB, 0, stream>>>(
      h3, W_up1, b_up1, h4, kB * kL, kUP, kR2, 0, 0, 0);
  cvt_f32_f16_kernel<<<(kB * kL * kUP + TB - 1) / TB, TB, 0, stream>>>(
      h4, h4f, (long)kB * kL * kUP);

  // fused up2 + final reduce (LDS-staged A, f16 operands)
  const size_t ldsBytes = (size_t)kL * kLdsRow * sizeof(_Float16);   // ~162 KB
  up2_fin_lds_wmma<<<dim3(kV / 16 / 8, kB, 1), TB, ldsBytes, stream>>>(
      h4f, Wuf, b_up2, W_fin, b_fin, out);
}
